// Agent_81466939670536
// MI455X (gfx1250) — matrix operand order repaired and checked by reading, not compiled, not run on hardware
//
#include <hip/hip_runtime.h>

typedef __attribute__((ext_vector_type(16))) __bf16 v16bf;
typedef __attribute__((ext_vector_type(8)))  float  v8f;
typedef __attribute__((ext_vector_type(4)))  unsigned int v4u;
typedef __attribute__((ext_vector_type(8)))  int v8i;
typedef __attribute__((ext_vector_type(4)))  int v4i;

#define HDIM 128
#define LSEG 512
#define WROW 272   /* 128 bf16 = 256B, padded to 272B -> conflict-free lane->bank mapping */

#define SM_H    0
#define SM_W    (LSEG * WROW)             /* 139264 */
#define SM_SC   (SM_W + HDIM * WROW)      /* 174080 */
#define SM_STOP (SM_SC + LSEG * 4)        /* 176128 */
#define SM_U    (SM_STOP + HDIM * 4)      /* 176640 */
#define SM_V    (SM_U + HDIM * 4)
#define SM_B1   (SM_V + HDIM * 4)
#define SM_B2   (SM_B1 + HDIM * 4)
#define SM_WSC  (SM_B2 + HDIM * 4)
#define SM_RED  (SM_WSC + HDIM * 4)       /* 179200 */
#define SM_TOTAL (SM_RED + 32 * 4)        /* 179328 bytes < 320KB WGP LDS */

/* Tensor Data Mover path: stage W2(bf16) -> LDS with native row padding.
   This toolchain exposes the 6-arg builtin (clang-23 / therock-10.0 arity):
   (uint32x4 g0, int32x8 g1, int32x4 g2, int32x4 g3, int32x8 g4, i32 cpol). */
#if defined(__HIP_DEVICE_COMPILE__) && __has_builtin(__builtin_amdgcn_tensor_load_to_lds) && __has_builtin(__builtin_amdgcn_s_wait_tensorcnt)
#define USE_TDM 1
#else
#define USE_TDM 0
#endif

union FragBF { uint4 q[2]; v16bf v; };

__device__ __forceinline__ unsigned f2bf1(float f) {
    unsigned u = __float_as_uint(f);
    return (u + 0x7FFFu + ((u >> 16) & 1u)) >> 16;   /* RNE f32->bf16 */
}
__device__ __forceinline__ unsigned packbf2(float lo, float hi) {
    return f2bf1(lo) | (f2bf1(hi) << 16);
}
__device__ __forceinline__ float swishf(float z) {
    return z / (1.0f + __expf(-z));
}

/* u = W1 @ w_seed, v = W1 @ w_node  (kills the first GEMM: rank-1 input) */
extern "C" __global__ void Agent_prep_uv(const float* __restrict__ W1,
                                         const float* __restrict__ wseed,
                                         const float* __restrict__ wnode,
                                         float* __restrict__ u, float* __restrict__ v) {
    int j = threadIdx.x;   /* 128 threads */
    float su = 0.f, sv = 0.f;
    #pragma unroll 8
    for (int k = 0; k < HDIM; ++k) {
        float w = W1[j * HDIM + k];
        su += w * wseed[k];
        sv += w * wnode[k];
    }
    u[j] = su; v[j] = sv;
}

/* W2 f32 -> bf16 row-major (B operand source: B[k][n] = W2[n][k]) */
extern "C" __global__ void Agent_prep_w2(const float* __restrict__ W2,
                                         unsigned short* __restrict__ w2bf) {
    int i = blockIdx.x * 256 + threadIdx.x;
    if (i < HDIM * HDIM) w2bf[i] = (unsigned short)f2bf1(W2[i]);
}

extern "C" __global__ __launch_bounds__(256)
void Agent_fused(const float* __restrict__ xs, const float* __restrict__ xn,
                 const float* __restrict__ b1, const float* __restrict__ b2,
                 const float* __restrict__ wscore, const float* __restrict__ Wstop,
                 const int* __restrict__ indptr,
                 const float* __restrict__ uvec, const float* __restrict__ vvec,
                 const unsigned short* __restrict__ w2bf,
                 float* __restrict__ out) {
    extern __shared__ char smem[];
    const int tid  = threadIdx.x;
    const int lane = tid & 31;
    const int wid  = tid >> 5;
    const int b    = blockIdx.x;

    const int start    = indptr[b * 3 + 0];
    const int mean_len = indptr[b * 3 + 1] - start;
    const int cand_len = indptr[b * 3 + 2] - start;

    float* sScores = (float*)(smem + SM_SC);
    float* sStop   = (float*)(smem + SM_STOP);
    float* sU      = (float*)(smem + SM_U);
    float* sV      = (float*)(smem + SM_V);
    float* sB1     = (float*)(smem + SM_B1);
    float* sB2     = (float*)(smem + SM_B2);
    float* sWsc    = (float*)(smem + SM_WSC);
    float* sRed    = (float*)(smem + SM_RED);

    /* ---- stage small vectors + zero stop accumulator ---- */
    if (tid < HDIM) {
        sU[tid]   = uvec[tid];
        sV[tid]   = vvec[tid];
        sB1[tid]  = b1[tid];
        sB2[tid]  = b2[tid];
        sWsc[tid] = wscore[tid];
        sStop[tid] = 0.f;
    }

#if USE_TDM
    /* ---- stage W2 bf16 via Tensor Data Mover: 2-D tensor of 4B elements,
       64 elems x 128 rows, whole-tensor tile, LDS padding of 4 DWORDs (16B)
       inserted every 64 DWORDs (256B = one row) => 272B padded rows. ---- */
    if (wid == 0) {
        unsigned long long ga = (unsigned long long)(uintptr_t)w2bf;
        unsigned ldsa = (unsigned)(uintptr_t)(void*)(smem + SM_W);
        v4u g0 = { 1u,                                   /* count=1, user mode      */
                   ldsa,                                 /* lds_addr  [63:32]       */
                   (unsigned)(ga & 0xFFFFFFFFu),         /* global_addr[95:64]      */
                   (unsigned)((ga >> 32) & 0x1FFFFFFu)   /* global_addr[120:96]     */
                   | (2u << 30) };                       /* type=2 ("image")        */
        v8i g1 = { (2 << 16)      /* data_size=4B           */
                 | (1 << 20)      /* pad_enable             */
                 | (5 << 22)      /* pad_interval: 256B     */
                 | (3 << 25),     /* pad_amount: 4 DWORDs   */
                   64 << 16,      /* tensor_dim0 = 64 elems (bits 79:48, low half)  */
                   128 << 16,     /* tensor_dim0 hi=0; tensor_dim1 = 128 (111:80)   */
                   64 << 16,      /* tensor_dim1 hi=0; tile_dim0 = 64 (127:112)     */
                   128,           /* tile_dim1 = 128; tile_dim2 = 0                 */
                   64,            /* tensor_dim0_stride = 64 elems (207:160 low)    */
                   0, 0 };        /* stride hi, tensor_dim1_stride = 0              */
        v4i gz4 = {};
        v8i gz8 = {};
        __builtin_amdgcn_tensor_load_to_lds(g0, g1, gz4, gz4, gz8, 0);
    }
#else
    /* ---- fallback: stage W2 bf16 into padded LDS rows (2048 uint4) ---- */
    {
        const uint4* src = (const uint4*)w2bf;
        for (int i = tid; i < (HDIM * HDIM * 2) / 16; i += 256) {
            int row = i >> 4, q = i & 15;
            *(uint4*)(smem + SM_W + row * WROW + q * 16) = src[i];
        }
    }
#endif

    /* ---- layer 1 (folded rank-1 GEMM) + swish -> bf16 h1 tile (overlaps TDM) ---- */
    for (int r = tid; r < LSEG; r += 256) {
        float a = xs[start + r];
        float c = xn[start + r];
        unsigned* dst = (unsigned*)(smem + SM_H + r * WROW);
        #pragma unroll 8
        for (int j = 0; j < HDIM; j += 2) {
            float z0 = fmaf(a, sU[j],     fmaf(c, sV[j],     sB1[j]));
            float z1 = fmaf(a, sU[j + 1], fmaf(c, sV[j + 1], sB1[j + 1]));
            dst[j >> 1] = packbf2(swishf(z0), swishf(z1));
        }
    }
#if USE_TDM
    __builtin_amdgcn_s_wait_tensorcnt(0);
#endif
    __syncthreads();

    /* ---- layer 2: bf16 WMMA GEMM fused with swish, score-dot, stop-mean ---- */
    const int hi  = lane >> 4;        /* 0: lanes 0-15, 1: lanes 16-31 */
    const int l15 = lane & 15;
    float stp[8];
    #pragma unroll
    for (int i = 0; i < 8; ++i) stp[i] = 0.f;

    for (int rt = wid * 4; rt < wid * 4 + 4; ++rt) {
        /* A fragments: 16x32 bf16 each; lane<16: K=kb..kb+7 & kb+16..kb+23,
           lane>=16: same rows, K shifted by 8 (ISA 7.12.2 layout) */
        FragBF A[4];
        const char* abase = smem + SM_H + (rt * 16 + l15) * WROW + hi * 16;
        #pragma unroll
        for (int ks = 0; ks < 4; ++ks) {
            A[ks].q[0] = *(const uint4*)(abase + ks * 64);
            A[ks].q[1] = *(const uint4*)(abase + ks * 64 + 32);
        }

        float sc[8];
        #pragma unroll
        for (int i = 0; i < 8; ++i) sc[i] = 0.f;

        #pragma unroll
        for (int ct = 0; ct < 8; ++ct) {
            const int n = ct * 16 + l15;           /* output column */
            v8f acc = {};                          /* C starts at 0; bias folded in epilogue */
            /* B fragment: 32x16 bf16; lane<16: col n, K=kb..kb+7 & kb+16..kb+23; lane>=16: K shifted by 8 */
            const char* bbase = smem + SM_W + n * WROW + hi * 16;
            #pragma unroll
            for (int ks = 0; ks < 4; ++ks) {
                FragBF Bf;
                Bf.q[0] = *(const uint4*)(bbase + ks * 64);
                Bf.q[1] = *(const uint4*)(bbase + ks * 64 + 32);
                acc = __builtin_amdgcn_wmma_f32_16x16x32_bf16(
                        false, A[ks].v, false, Bf.v, (short)0, acc, false, false);
            }
            /* epilogue straight from accumulators: +bias, swish, score/stop partials */
            const float bias = sB2[n];
            const float wsc  = sWsc[n];
            #pragma unroll
            for (int r8 = 0; r8 < 8; ++r8) {
                float h2 = swishf(acc[r8] + bias);
                sc[r8] += h2 * wsc;
                int grow = rt * 16 + hi * 8 + r8;
                stp[ct] += (grow < mean_len) ? h2 : 0.f;
            }
        }

        /* score reduce across the 16 lanes of each half (C tile: col = lane&15) */
        #pragma unroll
        for (int r8 = 0; r8 < 8; ++r8) {
            float s = sc[r8];
            s += __shfl_xor(s, 1, 32);
            s += __shfl_xor(s, 2, 32);
            s += __shfl_xor(s, 4, 32);
            s += __shfl_xor(s, 8, 32);
            sc[r8] = s;
        }
        if (l15 == 0) {
            #pragma unroll
            for (int r8 = 0; r8 < 8; ++r8)
                sScores[rt * 16 + hi * 8 + r8] = sc[r8];
        }
    }

    /* stop-mean partials -> LDS f32 atomics (ds_add_f32), per output column */
    #pragma unroll
    for (int ct = 0; ct < 8; ++ct)
        atomicAdd(&sStop[ct * 16 + l15], stp[ct]);
    __syncthreads();

    /* ---- masked log-softmax over 512 scores (2 per thread) ---- */
    const float NEGINF = -__builtin_inff();
    float a0 = (tid       < cand_len) ? sScores[tid]       : NEGINF;
    float a1 = (tid + 256 < cand_len) ? sScores[tid + 256] : NEGINF;

    float m = fmaxf(a0, a1);
    #pragma unroll
    for (int off = 16; off >= 1; off >>= 1) m = fmaxf(m, __shfl_xor(m, off, 32));
    if (lane == 0) sRed[wid] = m;
    __syncthreads();
    if (tid == 0) {
        float mm = sRed[0];
        #pragma unroll
        for (int i = 1; i < 8; ++i) mm = fmaxf(mm, sRed[i]);
        sRed[16] = mm;
    }
    __syncthreads();
    const float M = sRed[16];

    float e = __expf(a0 - M) + __expf(a1 - M);
    #pragma unroll
    for (int off = 16; off >= 1; off >>= 1) e += __shfl_xor(e, off, 32);
    if (lane == 0) sRed[8 + wid] = e;
    __syncthreads();
    if (tid == 0) {
        float s = 0.f;
        #pragma unroll
        for (int i = 0; i < 8; ++i) s += sRed[8 + i];
        sRed[17] = __logf(s);
        /* stop head: 2x128 dot + 2-way log-softmax */
        float inv = 1.0f / (float)(mean_len > 0 ? mean_len : 1);
        float t0 = 0.f, t1 = 0.f;
        for (int c = 0; c < HDIM; ++c) {
            float sn = sStop[c] * inv;
            t0 += sn * Wstop[c];
            t1 += sn * Wstop[HDIM + c];
        }
        float mm  = fmaxf(t0, t1);
        float lse = mm + __logf(__expf(t0 - mm) + __expf(t1 - mm));
        sRed[18] = t0 - lse;
        sRed[19] = t1 - lse;
    }
    __syncthreads();
    const float logS = sRed[17];
    const float sl0  = sRed[18];
    const float sl1  = sRed[19];

    float* orow = out + (size_t)b * (LSEG + 1);
    orow[tid]       = (a0 - M - logS) + sl0;
    orow[tid + 256] = (a1 - M - logS) + sl0;
    if (tid == 0) orow[LSEG] = sl1;
}

extern "C" void kernel_launch(void* const* d_in, const int* in_sizes, int n_in,
                              void* d_out, int out_size, void* d_ws, size_t ws_size,
                              hipStream_t stream) {
    const float* xs     = (const float*)d_in[0];
    const float* xn     = (const float*)d_in[1];
    const float* wseed  = (const float*)d_in[2];
    const float* wnode  = (const float*)d_in[3];
    const float* W1     = (const float*)d_in[4];
    const float* b1     = (const float*)d_in[5];
    const float* W2     = (const float*)d_in[6];
    const float* b2     = (const float*)d_in[7];
    const float* wscore = (const float*)d_in[8];
    const float* Wstop  = (const float*)d_in[9];
    const int*   indptr = (const int*)d_in[10];

    const int nB = in_sizes[10] / 3;   /* 2048 */

    float* u = (float*)d_ws;
    float* v = u + HDIM;
    unsigned short* w2bf = (unsigned short*)(v + HDIM);

    Agent_prep_uv<<<1, HDIM, 0, stream>>>(W1, wseed, wnode, u, v);
    Agent_prep_w2<<<(HDIM * HDIM + 255) / 256, 256, 0, stream>>>(W2, w2bf);
    Agent_fused<<<nB, 256, SM_TOTAL, stream>>>(xs, xn, b1, b2, wscore, Wstop,
                                               indptr, u, v, w2bf, (float*)d_out);
}
